// ASTCompressionRestoration_326417515243
// MI455X (gfx1250) — matrix operand order repaired and checked by reading, not compiled, not run on hardware
//
#include <hip/hip_runtime.h>

typedef __attribute__((ext_vector_type(16))) _Float16 v16h;
typedef __attribute__((ext_vector_type(8)))  _Float16 v8h;
typedef __attribute__((ext_vector_type(8)))  float    v8f;

// activation: 0 = none, 1 = relu, 2 = sigmoid
__device__ __forceinline__ float actf(float v, int act) {
    if (act == 1) return fmaxf(v, 0.f);
    if (act == 2) return 1.f / (1.f + __expf(-v));
    return v;
}

__device__ __forceinline__ v8h cvt8(float4 x, float4 y) {
    v8h h;
    h[0] = (_Float16)x.x; h[1] = (_Float16)x.y;
    h[2] = (_Float16)x.z; h[3] = (_Float16)x.w;
    h[4] = (_Float16)y.x; h[5] = (_Float16)y.y;
    h[6] = (_Float16)y.z; h[7] = (_Float16)y.w;
    return h;
}

__device__ __forceinline__ v8h zero8h() {
    v8h h;
#pragma unroll
    for (int j = 0; j < 8; ++j) h[j] = (_Float16)0.f;
    return h;
}

// ---------------------------------------------------------------------------
// WMMA GEMM:  out[M,N] = act(A[M,K] @ B + bias) (+ res)
//   btrans=1: B given as W[N,K];  btrans=0: B given as [K,N]
// REQUIRES: M % 128 == 0, N % 64 == 0, K % 32 == 0 (true for all call sites).
// Block = 256 thr = 8 waves; block tile 128x64; wave tile 32x32
// (4 WMMA/step with A/B fragment cross-reuse). Software-pipelined prefetch.
// ---------------------------------------------------------------------------
__global__ __launch_bounds__(256) void k_gemm_wmma(
    const float* __restrict__ A, const float* __restrict__ Bm,
    const float* __restrict__ bias, const float* __restrict__ res,
    float* __restrict__ out,
    int M, int N, int K, int btrans, int act,
    long sA, long sB, long sC)
{
    (void)M;
    A   += (long)blockIdx.z * sA;
    Bm  += (long)blockIdx.z * sB;
    out += (long)blockIdx.z * sC;
    if (res) res += (long)blockIdx.z * sC;

    __shared__ _Float16 As[128 * 40];
    __shared__ _Float16 Bs[64 * 40];

    const int tid  = threadIdx.x;
    const int lane = tid & 31, wv = tid >> 5;
    const int mw = wv & 3, nw = wv >> 2;          // 4x2 wave grid of 32x32 tiles
    const int l16 = lane & 15, hi = lane >> 4;
    const int row0 = blockIdx.x * 128, col0 = blockIdx.y * 64;

    // staging assignment: each thread owns 8 consecutive K (or N) elements
    const int ar = tid >> 2, ac8 = (tid & 3) * 8;     // A rows ar, ar+64
    const int tk = tid >> 3, tn8 = (tid & 7) * 8;     // B normal: k, n offset

    const float* Arow0 = A + (long)(row0 + ar) * K + ac8;
    const float* Arow1 = Arow0 + (long)64 * K;
    const float* Brow  = btrans ? (Bm + (long)(col0 + ar) * K + ac8)
                                : (Bm + (long)tk * N + col0 + tn8);
    const long   Bstep = btrans ? 32 : (long)32 * N;

    v8f acc00, acc01, acc10, acc11;
#pragma unroll
    for (int i = 0; i < 8; ++i) {
        acc00[i] = 0.f; acc01[i] = 0.f; acc10[i] = 0.f; acc11[i] = 0.f;
    }

    float4 a0 = *(const float4*)(Arow0);
    float4 a1 = *(const float4*)(Arow0 + 4);
    float4 a2 = *(const float4*)(Arow1);
    float4 a3 = *(const float4*)(Arow1 + 4);
    float4 b0 = *(const float4*)(Brow);
    float4 b1 = *(const float4*)(Brow + 4);

    int k0 = 0;
    while (true) {
        // commit staged regs to LDS (f32 -> f16)
        *(v8h*)&As[ar * 40 + ac8]        = cvt8(a0, a1);
        *(v8h*)&As[(ar + 64) * 40 + ac8] = cvt8(a2, a3);
        if (btrans) {
            *(v8h*)&Bs[ar * 40 + ac8] = cvt8(b0, b1);
        } else {
            v8h h = cvt8(b0, b1);
#pragma unroll
            for (int j = 0; j < 8; ++j) Bs[(tn8 + j) * 40 + tk] = h[j];
        }
        __syncthreads();

        const bool more = (k0 + 32) < K;
        if (more) {   // prefetch next chunk while WMMAs run
            Arow0 += 32; Arow1 += 32; Brow += Bstep;
            a0 = *(const float4*)(Arow0);
            a1 = *(const float4*)(Arow0 + 4);
            a2 = *(const float4*)(Arow1);
            a3 = *(const float4*)(Arow1 + 4);
            b0 = *(const float4*)(Brow);
            b1 = *(const float4*)(Brow + 4);
        }

        // fragments (ISA 16-bit A layout: lane group selects K halves)
        const _Float16* ap0 = &As[(mw * 32 + l16) * 40 + hi * 8];
        v8h x0 = *(const v8h*)ap0, x1 = *(const v8h*)(ap0 + 16);
        const _Float16* ap1 = ap0 + 16 * 40;
        v8h x2 = *(const v8h*)ap1, x3 = *(const v8h*)(ap1 + 16);
        const _Float16* bp0 = &Bs[(nw * 32 + l16) * 40 + hi * 8];
        v8h y0 = *(const v8h*)bp0, y1 = *(const v8h*)(bp0 + 16);
        const _Float16* bp1 = bp0 + 16 * 40;
        v8h z0 = *(const v8h*)bp1, z1 = *(const v8h*)(bp1 + 16);
        v16h af0, af1, bf0, bf1;
#pragma unroll
        for (int j = 0; j < 8; ++j) {
            af0[j] = x0[j]; af0[j + 8] = x1[j];
            af1[j] = x2[j]; af1[j + 8] = x3[j];
            bf0[j] = y0[j]; bf0[j + 8] = y1[j];
            bf1[j] = z0[j]; bf1[j + 8] = z1[j];
        }
        acc00 = __builtin_amdgcn_wmma_f32_16x16x32_f16(
            false, af0, false, bf0, (short)0, acc00, false, false);
        acc01 = __builtin_amdgcn_wmma_f32_16x16x32_f16(
            false, af0, false, bf1, (short)0, acc01, false, false);
        acc10 = __builtin_amdgcn_wmma_f32_16x16x32_f16(
            false, af1, false, bf0, (short)0, acc10, false, false);
        acc11 = __builtin_amdgcn_wmma_f32_16x16x32_f16(
            false, af1, false, bf1, (short)0, acc11, false, false);
        __syncthreads();
        if (!more) break;
        k0 += 32;
    }

    const int rowe0 = row0 + mw * 32 + hi * 8;
    const int rowe1 = rowe0 + 16;
    const int c0e   = col0 + nw * 32 + l16;
    const float bv0 = bias ? bias[c0e] : 0.f;
    const float bv1 = bias ? bias[c0e + 16] : 0.f;
#pragma unroll
    for (int r = 0; r < 8; ++r) {
        const long ro0 = (long)(rowe0 + r) * N;
        const long ro1 = (long)(rowe1 + r) * N;
        float v00 = actf(acc00[r] + bv0, act);
        float v01 = actf(acc01[r] + bv1, act);
        float v10 = actf(acc10[r] + bv0, act);
        float v11 = actf(acc11[r] + bv1, act);
        if (res) {
            v00 += res[ro0 + c0e]; v01 += res[ro0 + c0e + 16];
            v10 += res[ro1 + c0e]; v11 += res[ro1 + c0e + 16];
        }
        out[ro0 + c0e]      = v00;
        out[ro0 + c0e + 16] = v01;
        out[ro1 + c0e]      = v10;
        out[ro1 + c0e + 16] = v11;
    }
}

// ---------------------------------------------------------------------------
// Implicit-GEMM 3x3 conv on token (NHWC) layout via WMMA.
// in: [B, IH*IW, Cin], Wt: [Cout, Cin, 3, 3]. Block tile 64x64, 2 WMMA/step.
// Mtot % 64 == 0 always; Cout may be small (3) -> column guards kept.
// ---------------------------------------------------------------------------
__global__ __launch_bounds__(256) void k_conv3_wmma(
    const float* __restrict__ in, const float* __restrict__ Wt,
    const float* __restrict__ bias, const float* __restrict__ res,
    float* __restrict__ out,
    int Mtot, int Cin, int Cout,
    int iw_l2, int ow_l2, int strd, int pad, int dil, int act,
    int out_cs, int out_c0)
{
    (void)Mtot;
    __shared__ _Float16 As[64 * 40];
    __shared__ _Float16 Bs[64 * 40];

    const int tid  = threadIdx.x;
    const int lane = tid & 31, wv = tid >> 5;
    const int mw = wv & 3, nw = wv >> 2;
    const int l16 = lane & 15, hi = lane >> 4;
    const int row0 = blockIdx.x * 64, col0 = blockIdx.y * 64;
    const int IW = 1 << iw_l2, OW = 1 << ow_l2;
    const int ohw_l2 = 2 * ow_l2, ihw_l2 = 2 * iw_l2;

    const int ar = tid >> 2, ac8 = (tid & 3) * 8;
    const int arow = row0 + ar;
    const int ab  = arow >> ohw_l2;
    const int aop = arow & ((1 << ohw_l2) - 1);
    const int aoy = aop >> ow_l2, aox = aop & (OW - 1);
    const int co_b = col0 + ar;
    const bool cfull = (Cin & 31) == 0;

    v8f acc0, acc1;
#pragma unroll
    for (int i = 0; i < 8; ++i) { acc0[i] = 0.f; acc1[i] = 0.f; }

    const int nch = (Cin + 31) >> 5;
    for (int tap = 0; tap < 9; ++tap) {
        const int ky = tap / 3, kx = tap - ky * 3;
        const int iy = aoy * strd - pad + ky * dil;
        const int ix = aox * strd - pad + kx * dil;
        const bool sv = (unsigned)iy < (unsigned)IW && (unsigned)ix < (unsigned)IW;
        const float* ipt = in + ((long)((ab << ihw_l2) + (iy << iw_l2) + ix)) * Cin;

        for (int cc = 0; cc < nch; ++cc) {
            const int c0 = cc << 5;
            // A tile
            v8h ha;
            if (cfull) {
                if (sv) {
                    float4 f0 = *(const float4*)(ipt + c0 + ac8);
                    float4 f1 = *(const float4*)(ipt + c0 + ac8 + 4);
                    ha = cvt8(f0, f1);
                } else ha = zero8h();
            } else {
                ha = zero8h();
                if (sv) {
#pragma unroll
                    for (int j = 0; j < 8; ++j) {
                        int ci = c0 + ac8 + j;
                        if (ci < Cin) ha[j] = (_Float16)ipt[ci];
                    }
                }
            }
            *(v8h*)&As[ar * 40 + ac8] = ha;
            // B tile (weights for this tap)
            v8h hb = zero8h();
            if (co_b < Cout) {
#pragma unroll
                for (int j = 0; j < 8; ++j) {
                    int ci = c0 + ac8 + j;
                    if (ci < Cin)
                        hb[j] = (_Float16)Wt[((long)(co_b * Cin + ci)) * 9 + tap];
                }
            }
            *(v8h*)&Bs[ar * 40 + ac8] = hb;
            __syncthreads();

            const _Float16* ap = &As[(mw * 16 + l16) * 40 + hi * 8];
            v8h x0 = *(const v8h*)ap, x1 = *(const v8h*)(ap + 16);
            const _Float16* bp0 = &Bs[(nw * 32 + l16) * 40 + hi * 8];
            v8h y0 = *(const v8h*)bp0, y1 = *(const v8h*)(bp0 + 16);
            const _Float16* bp1 = bp0 + 16 * 40;
            v8h z0 = *(const v8h*)bp1, z1 = *(const v8h*)(bp1 + 16);
            v16h af, bf0, bf1;
#pragma unroll
            for (int j = 0; j < 8; ++j) {
                af[j]  = x0[j]; af[j + 8]  = x1[j];
                bf0[j] = y0[j]; bf0[j + 8] = y1[j];
                bf1[j] = z0[j]; bf1[j + 8] = z1[j];
            }
            acc0 = __builtin_amdgcn_wmma_f32_16x16x32_f16(
                false, af, false, bf0, (short)0, acc0, false, false);
            acc1 = __builtin_amdgcn_wmma_f32_16x16x32_f16(
                false, af, false, bf1, (short)0, acc1, false, false);
            __syncthreads();
        }
    }

    const int rowe = row0 + mw * 16 + hi * 8;
    const int c0e  = col0 + nw * 32 + l16;
    const bool g0 = c0e < Cout, g1 = (c0e + 16) < Cout;
    const float bv0 = (bias && g0) ? bias[c0e] : 0.f;
    const float bv1 = (bias && g1) ? bias[c0e + 16] : 0.f;
#pragma unroll
    for (int r = 0; r < 8; ++r) {
        const long ro = (long)(rowe + r) * out_cs + out_c0;
        if (g0) {
            float v = actf(acc0[r] + bv0, act);
            if (res) v += res[ro + c0e];
            out[ro + c0e] = v;
        }
        if (g1) {
            float v = actf(acc1[r] + bv1, act);
            if (res) v += res[ro + c0e + 16];
            out[ro + c0e + 16] = v;
        }
    }
}

// ---------------------------------------------------------------------------
// conv_transpose2d via WMMA implicit GEMM (lhs_dilation 2, pad (1,2), k=3):
// in [B,64*64,128] tokens -> out [B,128*128,64] tokens, ReLU.
// wt[o,i,ky,kx] = w[i,o,2-ky,2-kx].  Block tile 64x64 (N=64 exactly).
// ---------------------------------------------------------------------------
__global__ __launch_bounds__(256) void k_convT_wmma(
    const float* __restrict__ in, const float* __restrict__ Wt,
    const float* __restrict__ bias, float* __restrict__ out)
{
    __shared__ _Float16 As[64 * 40];
    __shared__ _Float16 Bs[64 * 40];

    const int tid  = threadIdx.x;
    const int lane = tid & 31, wv = tid >> 5;
    const int mw = wv & 3, nw = wv >> 2;
    const int l16 = lane & 15, hi = lane >> 4;
    const int row0 = blockIdx.x * 64;

    const int ar = tid >> 2, ac8 = (tid & 3) * 8;
    const int arow = row0 + ar;
    const int ab = arow >> 14, aop = arow & 16383;
    const int aoy = aop >> 7, aox = aop & 127;
    const int co_b = ar;   // 64 out channels

    v8f acc0, acc1;
#pragma unroll
    for (int i = 0; i < 8; ++i) { acc0[i] = 0.f; acc1[i] = 0.f; }

    for (int tap = 0; tap < 9; ++tap) {
        const int ky = tap / 3, kx = tap - ky * 3;
        const int dy = aoy + ky - 1, dx = aox + kx - 1;
        const int iy = dy >> 1, ix = dx >> 1;
        const bool sv = !(dy & 1) && !(dx & 1) &&
                        (unsigned)iy < 64u && (unsigned)ix < 64u;
        const float* ipt = in + ((long)(ab * 4096 + iy * 64 + ix)) * 128;
        const int wk = 8 - tap;   // flipped tap

        for (int cc = 0; cc < 4; ++cc) {
            const int c0 = cc << 5;
            v8h ha;
            if (sv) {
                float4 f0 = *(const float4*)(ipt + c0 + ac8);
                float4 f1 = *(const float4*)(ipt + c0 + ac8 + 4);
                ha = cvt8(f0, f1);
            } else ha = zero8h();
            *(v8h*)&As[ar * 40 + ac8] = ha;

            v8h hb;
#pragma unroll
            for (int j = 0; j < 8; ++j) {
                int ci = c0 + ac8 + j;
                hb[j] = (_Float16)Wt[((long)(ci * 64 + co_b)) * 9 + wk];
            }
            *(v8h*)&Bs[ar * 40 + ac8] = hb;
            __syncthreads();

            const _Float16* ap = &As[(mw * 16 + l16) * 40 + hi * 8];
            v8h x0 = *(const v8h*)ap, x1 = *(const v8h*)(ap + 16);
            const _Float16* bp0 = &Bs[(nw * 32 + l16) * 40 + hi * 8];
            v8h y0 = *(const v8h*)bp0, y1 = *(const v8h*)(bp0 + 16);
            const _Float16* bp1 = bp0 + 16 * 40;
            v8h z0 = *(const v8h*)bp1, z1 = *(const v8h*)(bp1 + 16);
            v16h af, bf0, bf1;
#pragma unroll
            for (int j = 0; j < 8; ++j) {
                af[j]  = x0[j]; af[j + 8]  = x1[j];
                bf0[j] = y0[j]; bf0[j + 8] = y1[j];
                bf1[j] = z0[j]; bf1[j + 8] = z1[j];
            }
            acc0 = __builtin_amdgcn_wmma_f32_16x16x32_f16(
                false, af, false, bf0, (short)0, acc0, false, false);
            acc1 = __builtin_amdgcn_wmma_f32_16x16x32_f16(
                false, af, false, bf1, (short)0, acc1, false, false);
            __syncthreads();
        }
    }

    const int rowe = row0 + mw * 16 + hi * 8;
    const int c0e  = nw * 32 + l16;
    const float bv0 = bias[c0e], bv1 = bias[c0e + 16];
#pragma unroll
    for (int r = 0; r < 8; ++r) {
        const long ro = (long)(rowe + r) * 64;
        out[ro + c0e]      = fmaxf(acc0[r] + bv0, 0.f);
        out[ro + c0e + 16] = fmaxf(acc1[r] + bv1, 0.f);
    }
}

// ---------------------------------------------------------------------------
// LayerNorm over channel dim, token layout [ntok, C]; one wave per token.
// ---------------------------------------------------------------------------
__global__ __launch_bounds__(256) void k_layernorm(
    const float* __restrict__ in, const float* __restrict__ g,
    const float* __restrict__ be, float* __restrict__ out, long ntok, int C)
{
    int wv = threadIdx.x >> 5;
    long tok = (long)blockIdx.x * 8 + wv;
    if (tok >= ntok) return;
    int lane = threadIdx.x & 31;
    const float* p = in + tok * C;
    float s = 0.f, s2 = 0.f;
    for (int c = lane; c < C; c += 32) { float v = p[c]; s += v; s2 += v * v; }
    for (int o = 16; o > 0; o >>= 1) {
        s  += __shfl_xor(s, o, 32);
        s2 += __shfl_xor(s2, o, 32);
    }
    float m = s / C;
    float var = s2 / C - m * m;
    float inv = rsqrtf(var + 1e-5f);
    float* q = out + tok * C;
    for (int c = lane; c < C; c += 32)
        q[c] = (p[c] - m) * inv * g[c] + be[c];
}

// ---------------------------------------------------------------------------
// Depthwise 3x3 pad=1 on first h channels of token buffer (row stride Cs).
// ---------------------------------------------------------------------------
__global__ __launch_bounds__(256) void k_dw3x3(
    const float* __restrict__ in, const float* __restrict__ w,
    const float* __restrict__ bias, float* __restrict__ out,
    int NT, int w_l2, int h_l2, int Cs)
{
    long idx = (long)blockIdx.x * 256 + threadIdx.x;
    long total = (long)NT << h_l2;
    if (idx >= total) return;
    int c = (int)(idx & ((1 << h_l2) - 1));
    long row = idx >> h_l2;
    int W = 1 << w_l2;
    int b = (int)(row >> (2 * w_l2));
    int pix = (int)(row & ((1 << (2 * w_l2)) - 1));
    int y = pix >> w_l2, x = pix & (W - 1);
    float acc = bias[c];
    for (int ky = 0; ky < 3; ++ky) {
        int iy = y + ky - 1;
        if ((unsigned)iy >= (unsigned)W) continue;
        for (int kx = 0; kx < 3; ++kx) {
            int ix = x + kx - 1;
            if ((unsigned)ix >= (unsigned)W) continue;
            long t = ((long)b << (2 * w_l2)) + (iy << w_l2) + ix;
            acc += in[t * Cs + c] * w[c * 9 + ky * 3 + kx];
        }
    }
    out[row * Cs + c] = acc;
}

// copy channel slice [c0..c0+2^cnt_l2) between token buffers
__global__ __launch_bounds__(256) void k_copy_slice(
    const float* __restrict__ in, float* __restrict__ out,
    int NT, int cs_in, int c0_in, int cs_out, int c0_out, int cnt_l2)
{
    long idx = (long)blockIdx.x * 256 + threadIdx.x;
    long total = (long)NT << cnt_l2;
    if (idx >= total) return;
    int c = (int)(idx & ((1 << cnt_l2) - 1));
    long row = idx >> cnt_l2;
    out[row * cs_out + c0_out + c] = in[row * cs_in + c0_in + c];
}

// attn = w1*softmax(S) + w2*relu(S)^2, in place; one block per row of 4096.
__global__ __launch_bounds__(256) void k_attn_combine(
    float* __restrict__ S, const float* __restrict__ w1p,
    const float* __restrict__ w2p, int L)
{
    long row = blockIdx.x;
    float* s = S + row * (long)L;
    __shared__ float buf[4096];
    __shared__ float red[256];
    int tid = threadIdx.x;
    float mx = -3.0e38f;
    for (int i = tid; i < L; i += 256) { float v = s[i]; buf[i] = v; mx = fmaxf(mx, v); }
    red[tid] = mx; __syncthreads();
    for (int o = 128; o > 0; o >>= 1) {
        if (tid < o) red[tid] = fmaxf(red[tid], red[tid + o]);
        __syncthreads();
    }
    mx = red[0]; __syncthreads();
    float sum = 0.f;
    for (int i = tid; i < L; i += 256) sum += __expf(buf[i] - mx);
    red[tid] = sum; __syncthreads();
    for (int o = 128; o > 0; o >>= 1) {
        if (tid < o) red[tid] += red[tid + o];
        __syncthreads();
    }
    sum = red[0];
    float e1 = __expf(w1p[0]), e2 = __expf(w2p[0]);
    float w1 = e1 / (e1 + e2), w2 = e2 / (e1 + e2);
    float inv = 1.f / sum;
    for (int i = tid; i < L; i += 256) {
        float v = buf[i];
        float sm = __expf(v - mx) * inv;
        float r = fmaxf(v, 0.f);
        s[i] = w1 * sm + w2 * r * r;
    }
}

// out = x + a * t, elementwise
__global__ __launch_bounds__(256) void k_fma(
    const float* __restrict__ x, const float* __restrict__ a,
    const float* __restrict__ t, float* __restrict__ out, long n)
{
    long i = (long)blockIdx.x * 256 + threadIdx.x;
    if (i < n) out[i] = x[i] + a[i] * t[i];
}

// NCHW (C=3) -> token layout [B, HW, 3]
__global__ __launch_bounds__(256) void k_tok_from_nchw(
    const float* __restrict__ in, float* __restrict__ out, int hw_l2, long total)
{
    long idx = (long)blockIdx.x * 256 + threadIdx.x;
    if (idx >= total) return;
    long n = idx & ((1 << hw_l2) - 1);
    int t = (int)(idx >> hw_l2);
    int b = t / 3, c = t - b * 3;
    out[(((long)b << hw_l2) + n) * 3 + c] = in[idx];
}

// class select + residual + clip; writes NCHW output directly
__global__ __launch_bounds__(256) void k_final(
    const float* __restrict__ x, const int* __restrict__ cid,
    const float* __restrict__ OJ, const float* __restrict__ O2,
    const float* __restrict__ OD, float* __restrict__ out, int hw_l2, long total)
{
    long idx = (long)blockIdx.x * 256 + threadIdx.x;
    if (idx >= total) return;
    long n = idx & ((1 << hw_l2) - 1);
    int t = (int)(idx >> hw_l2);
    int b = t / 3, c = t - b * 3;
    long o = (((long)b << hw_l2) + n) * 3 + c;
    int k = cid[b];
    float v = (k == 10) ? OJ[o] : (k == 9) ? O2[o] : OD[o];
    v += x[idx];
    out[idx] = fminf(fmaxf(v, 0.f), 1.f);
}

// ---------------------------------------------------------------------------
// Host orchestration
// ---------------------------------------------------------------------------
extern "C" void kernel_launch(void* const* d_in, const int* in_sizes, int n_in,
                              void* d_out, int out_size, void* d_ws, size_t ws_size,
                              hipStream_t stream)
{
    (void)in_sizes; (void)n_in; (void)out_size; (void)ws_size;
    const float* x  = (const float*)d_in[0];
    const int* cid  = (const int*)d_in[1];
#define PP(i) ((const float*)d_in[(i)])

    // input index map (depth-first dict-insertion order of setup_inputs params)
    enum {
        I_CONV1_W = 2, I_CONV1_B = 3,
        I_F1 = 4,
        I_CONV2_W = 14, I_CONV2_B = 15,
        I_F2 = 16,
        I_A1 = 26,
        I_F3 = 33,
        I_A2 = 43,
        I_F4 = 50,
        I_DC_W = 60, I_DC_B = 61,
        I_F5 = 62,
        I_JD = 72,
        I_JOUT_W = 90, I_JOUT_B = 91,
        I_TR = 92,
        I_KOUT_W = 98, I_KOUT_B = 99,
        I_DEF_W = 100, I_DEF_B = 101
    };

    const int B = 2, HW1 = 16384, NT1 = B * HW1, NT2 = B * 4096;

    float* ws = (float*)d_ws;
    size_t off = 0;
    auto alloc = [&](size_t n) { float* p = ws + off; off += n; return p; };
    float* xt = alloc((size_t)NT1 * 3);
    float* L[5]; for (int i = 0; i < 5; ++i) L[i] = alloc((size_t)NT1 * 64);
    float* U[5]; for (int i = 0; i < 5; ++i) U[i] = alloc((size_t)NT2 * 128);
    float* S  = alloc((size_t)B * 4096 * 4096);
    float* F  = alloc((size_t)NT1 * 192);
    float* OJ = alloc((size_t)NT1 * 3);
    float* O2 = alloc((size_t)NT1 * 3);
    float* OD = alloc((size_t)NT1 * 3);

    auto gemm = [&](const float* A, const float* Bm, const float* bias,
                    const float* res, float* out, int M, int N, int K,
                    int btrans, int act, int nb, long sA, long sB, long sC) {
        dim3 g((M + 127) / 128, (N + 63) / 64, nb);
        k_gemm_wmma<<<g, 256, 0, stream>>>(A, Bm, bias, res, out, M, N, K,
                                           btrans, act, sA, sB, sC);
    };
    auto conv3 = [&](const float* in, const float* W_, const float* b_,
                     const float* res, float* out, int Cin, int Cout,
                     int iw_l2, int ow_l2, int strd, int pad, int dil,
                     int act, int out_cs, int out_c0) {
        int Mtot = B << (2 * ow_l2);
        dim3 g((Mtot + 63) / 64, (Cout + 63) / 64, 1);
        k_conv3_wmma<<<g, 256, 0, stream>>>(in, W_, b_, res, out, Mtot, Cin, Cout,
                                            iw_l2, ow_l2, strd, pad, dil, act,
                                            out_cs, out_c0);
    };
    auto ln = [&](const float* in, int base, float* out, int NT, int C) {
        k_layernorm<<<(NT + 7) / 8, 256, 0, stream>>>(in, PP(base), PP(base + 1),
                                                      out, NT, C);
    };
    auto frfn = [&](const float* in, float* t1, float* t2, float* t3, float* out,
                    int base, int C, int w_l2) {
        int NT = B << (2 * w_l2);
        int h = C / 2, h_l2 = (C == 64) ? 5 : 6;
        ln(in, base, t1, NT, C);
        gemm(t1, PP(base + 2), PP(base + 3), nullptr, t2, NT, C, C, 1, 0, 1, 0, 0, 0);
        gemm(t2, PP(base + 4), PP(base + 5), nullptr, t1, NT, C, C, 1, 0, 1, 0, 0, 0);
        long ndw = (long)NT * h;
        k_dw3x3<<<(int)((ndw + 255) / 256), 256, 0, stream>>>(
            t1, PP(base + 6), PP(base + 7), t3, NT, w_l2, h_l2, C);
        k_copy_slice<<<(int)((ndw + 255) / 256), 256, 0, stream>>>(
            t1, t3, NT, C, h, C, h, h_l2);
        gemm(t3, PP(base + 8), PP(base + 9), in, out, NT, C, C, 1, 0, 1, 0, 0, 0);
    };
    auto assa = [&](const float* in, float* t, float* Q, float* Kk, float* V,
                    float* out, int base) {
        ln(in, base, t, NT2, 128);
        gemm(t, PP(base + 2), nullptr, nullptr, Q,  NT2, 128, 128, 1, 0, 1, 0, 0, 0);
        gemm(t, PP(base + 3), nullptr, nullptr, Kk, NT2, 128, 128, 1, 0, 1, 0, 0, 0);
        gemm(t, PP(base + 4), nullptr, nullptr, V,  NT2, 128, 128, 1, 0, 1, 0, 0, 0);
        long sQ = (long)4096 * 128, sS = (long)4096 * 4096;
        gemm(Q, Kk, nullptr, nullptr, S, 4096, 4096, 128, 1, 0, B, sQ, sQ, sS);
        k_attn_combine<<<NT2, 256, 0, stream>>>(S, PP(base + 5), PP(base + 6), 4096);
        gemm(S, V, nullptr, in, out, 4096, 128, 4096, 0, 0, B, sS, sQ, sQ);
    };

    // ---- forward pass ----
    {
        long n = (long)B * 3 * HW1;
        k_tok_from_nchw<<<(int)((n + 255) / 256), 256, 0, stream>>>(x, xt, 14, n);
    }
    conv3(xt, PP(I_CONV1_W), PP(I_CONV1_B), nullptr, L[0], 3, 64, 7, 7, 1, 1, 1, 1, 64, 0);
    frfn(L[0], L[1], L[2], L[3], L[4], I_F1, 64, 7);
    conv3(L[4], PP(I_CONV2_W), PP(I_CONV2_B), nullptr, U[0], 64, 128, 7, 6, 2, 1, 1, 1, 128, 0);
    frfn(U[0], U[1], U[2], U[3], U[4], I_F2, 128, 6);
    assa(U[4], U[0], U[1], U[2], U[3], U[0], I_A1);
    frfn(U[0], U[1], U[2], U[3], U[4], I_F3, 128, 6);
    assa(U[4], U[0], U[1], U[2], U[3], U[0], I_A2);
    frfn(U[0], U[1], U[2], U[3], U[4], I_F4, 128, 6);
    k_convT_wmma<<<dim3(NT1 / 64, 1, 1), 256, 0, stream>>>(U[4], PP(I_DC_W),
                                                           PP(I_DC_B), L[0]);
    frfn(L[0], L[1], L[2], L[3], L[4], I_F5, 64, 7);
    const float* x5 = L[4];

    // jpeg head: DCSC
    conv3(x5, PP(I_JD + 0), PP(I_JD + 1), nullptr, F, 64, 64, 7, 7, 1, 1, 1, 0, 192, 0);
    conv3(x5, PP(I_JD + 2), PP(I_JD + 3), nullptr, F, 64, 64, 7, 7, 1, 2, 2, 0, 192, 64);
    conv3(x5, PP(I_JD + 4), PP(I_JD + 5), nullptr, F, 64, 64, 7, 7, 1, 4, 4, 0, 192, 128);
    gemm(F, PP(I_JD + 6), PP(I_JD + 7), nullptr, L[0], NT1, 64, 192, 1, 0, 1, 0, 0, 0);
    conv3(L[0], PP(I_JD + 8),  PP(I_JD + 9),  nullptr, L[1], 64, 64, 7, 7, 1, 1, 1, 1, 64, 0);
    conv3(L[1], PP(I_JD + 10), PP(I_JD + 11), nullptr, L[0], 64, 64, 7, 7, 1, 1, 1, 1, 64, 0);
    conv3(L[0], PP(I_JD + 12), PP(I_JD + 13), nullptr, L[1], 64, 64, 7, 7, 1, 1, 1, 1, 64, 0);
    conv3(L[1], PP(I_JD + 14), PP(I_JD + 15), nullptr, L[0], 64, 64, 7, 7, 1, 1, 1, 1, 64, 0);
    conv3(L[0], PP(I_JD + 16), PP(I_JD + 17), x5,      L[1], 64, 64, 7, 7, 1, 1, 1, 0, 64, 0);
    conv3(L[1], PP(I_JOUT_W), PP(I_JOUT_B), nullptr, OJ, 64, 3, 7, 7, 1, 1, 1, 0, 3, 0);

    // j2k head: TexRef
    gemm(x5, PP(I_TR + 0), PP(I_TR + 1), nullptr, L[0], NT1, 64, 64, 1, 2, 1, 0, 0, 0);
    conv3(x5,  PP(I_TR + 2), PP(I_TR + 3), nullptr, L[1], 64, 64, 7, 7, 1, 1, 1, 1, 64, 0);
    conv3(L[1], PP(I_TR + 4), PP(I_TR + 5), nullptr, L[2], 64, 64, 7, 7, 1, 1, 1, 0, 64, 0);
    {
        long n = (long)NT1 * 64;
        k_fma<<<(int)((n + 255) / 256), 256, 0, stream>>>(x5, L[0], L[2], L[1], n);
    }
    conv3(L[1], PP(I_KOUT_W), PP(I_KOUT_B), nullptr, O2, 64, 3, 7, 7, 1, 1, 1, 0, 3, 0);

    // default head
    conv3(x5, PP(I_DEF_W), PP(I_DEF_B), nullptr, OD, 64, 3, 7, 7, 1, 1, 1, 0, 3, 0);

    // select + residual + clip
    {
        long n = (long)B * 3 * HW1;
        k_final<<<(int)((n + 255) / 256), 256, 0, stream>>>(x, cid, OJ, O2, OD,
                                                            (float*)d_out, 14, n);
    }
#undef PP
}
